// Attention_48498770706573
// MI455X (gfx1250) — matrix operand order repaired and checked by reading, not compiled, not run on hardware
//
#include <hip/hip_runtime.h>
#include <stdint.h>

// ---------------- problem constants (from reference) ----------------
#define BB  2
#define NQS 2048
#define NKS 2048
#define DD  1024
#define NH  16
#define HD  64            // head dim

typedef __attribute__((ext_vector_type(16))) __bf16 v16bf;
typedef __attribute__((ext_vector_type(2)))  __bf16 v2bf;
typedef __attribute__((ext_vector_type(8)))  float  v8f;
typedef __attribute__((ext_vector_type(2)))  float  v2f;
typedef __attribute__((ext_vector_type(4)))  unsigned int v4u;
typedef __attribute__((ext_vector_type(8)))  int v8i;
typedef __attribute__((ext_vector_type(4)))  int v4i;

union FragBF {
    v16bf v;
    uint4 q[2];
};

static __device__ __forceinline__ v8f zero8() {
    v8f z;
#pragma unroll
    for (int i = 0; i < 8; ++i) z[i] = 0.0f;
    return z;
}

// fp32 -> bf16 (hardware v_cvt_bf16)
static __device__ __forceinline__ unsigned short f2bf(float x) {
    union { __bf16 h; unsigned short u; } c;
    c.h = (__bf16)x;
    return c.u;
}
// fp32 pair -> packed bf16x2: single v_cvt_pk_bf16_f32 via vector fptrunc
static __device__ __forceinline__ unsigned pack2(float lo, float hi) {
    v2f f;
    f[0] = lo;
    f[1] = hi;
    union { v2bf v; unsigned u; } c;
    c.v = __builtin_convertvector(f, v2bf);
    return c.u;
}

// =====================================================================
// Tensor Data Mover: 2-D bf16 tile  global -> LDS  (CDNA5 TDM, D# per
// CDNA5 ISA).  Tracked by TENSORcnt.
// 6-arg builtin on this toolchain: (v4u g0, v8i g1, v4i, v4i, v8i, cpol)
// =====================================================================
#if __has_builtin(__builtin_amdgcn_tensor_load_to_lds)
#define HAVE_TDM 1
static __device__ __forceinline__ void tdm_load_2d_bf16(
    unsigned lds_off, const void* gptr,
    unsigned tile_w, unsigned tile_h, unsigned row_stride_elems)
{
    unsigned long long ga = (unsigned long long)gptr;
    v4u g0;
    g0[0] = 1u;                                           // count=1, user-mode
    g0[1] = lds_off;                                      // lds_addr[31:0]
    g0[2] = (unsigned)(ga & 0xffffffffu);                 // global_addr[31:0]
    g0[3] = (unsigned)((ga >> 32) & 0x1ffffffu)           // global_addr[56:32]
            | 0x80000000u;                                // type=2 ("image")
    v8i g1;
    g1[0] = (int)(1u << 16);                              // wg_mask=0, data_size=1 (2B)
    g1[1] = (int)((tile_w & 0xffffu) << 16);              // tensor_dim0[15:0]
    g1[2] = (int)(((tile_w >> 16) & 0xffffu)
            | ((tile_h & 0xffffu) << 16));                // tdim0[31:16], tdim1[15:0]
    g1[3] = (int)(((tile_h >> 16) & 0xffffu)
            | ((tile_w & 0xffffu) << 16));                // tdim1[31:16], tile_dim0
    g1[4] = (int)(tile_h & 0xffffu);                      // tile_dim1, tile_dim2=0
    g1[5] = (int)row_stride_elems;                        // tensor_dim0_stride[31:0]
    g1[6] = 0;                                            // stride[47:32], d1stride lo
    g1[7] = 0;
    v4i z4 = {0, 0, 0, 0};
    v8i z8 = {0, 0, 0, 0, 0, 0, 0, 0};
    __builtin_amdgcn_tensor_load_to_lds(g0, g1, z4, z4, z8, 0); // cpol=0
}
#endif

static __device__ __forceinline__ void wait_tensorcnt0() {
#if __has_builtin(__builtin_amdgcn_s_wait_tensorcnt)
    __builtin_amdgcn_s_wait_tensorcnt(0);
#else
    asm volatile("s_wait_tensorcnt 0" ::: "memory");
#endif
}

// =====================================================================
// Projection GEMM:  out = (X @ W^T + bias) * scale     (bf16 output)
//   Pipelined: next fp32 tile loads into registers while WMMAs consume
//   the current LDS tile (bf16).  128x128 tile, 256 thr = 8 waves.
// =====================================================================
__global__ __launch_bounds__(256)
void proj_gemm(const float* __restrict__ X,
               const float* __restrict__ W,
               const float* __restrict__ bias,
               unsigned short* __restrict__ out,
               float scale, int transposeV)
{
    __shared__ __align__(16) unsigned short lA[128 * 32];
    __shared__ __align__(16) unsigned short lB[128 * 32];

    const int tid  = threadIdx.x;
    const int lane = tid & 31;
    const int wave = tid >> 5;
    const int rowBase = blockIdx.x * 128;
    const int colBase = blockIdx.y * 128;
    const int wm = (wave & 3) * 32;
    const int wn = (wave >> 2) * 64;

    v8f acc[2][4];
#pragma unroll
    for (int mt = 0; mt < 2; ++mt)
#pragma unroll
        for (int nt = 0; nt < 4; ++nt) acc[mt][nt] = zero8();

    float4 ra[4], rb[4];

    // prologue: load tile k0=0 into registers
#pragma unroll
    for (int i = 0; i < 4; ++i) {
        int idx = tid + i * 256;
        int r = idx >> 3, c = (idx & 7) * 4;
        ra[i] = *(const float4*)(X + (size_t)(rowBase + r) * DD + c);
        rb[i] = *(const float4*)(W + (size_t)(colBase + r) * DD + c);
    }

    for (int k0 = 0; k0 < DD; k0 += 32) {
        // ---- commit staged registers to LDS as bf16 ----
#pragma unroll
        for (int i = 0; i < 4; ++i) {
            int idx = tid + i * 256;
            int r = idx >> 3, c = (idx & 7) * 4;
            *(uint2*)(lA + r * 32 + c) = make_uint2(pack2(ra[i].x, ra[i].y), pack2(ra[i].z, ra[i].w));
            *(uint2*)(lB + r * 32 + c) = make_uint2(pack2(rb[i].x, rb[i].y), pack2(rb[i].z, rb[i].w));
        }
        __syncthreads();

        // ---- issue next tile's global loads (overlap with WMMA below) ----
        if (k0 + 32 < DD) {
#pragma unroll
            for (int i = 0; i < 4; ++i) {
                int idx = tid + i * 256;
                int r = idx >> 3, c = (idx & 7) * 4;
                ra[i] = *(const float4*)(X + (size_t)(rowBase + r) * DD + k0 + 32 + c);
                rb[i] = *(const float4*)(W + (size_t)(colBase + r) * DD + k0 + 32 + c);
            }
        }

        // ---- fragments + WMMA ----
        FragBF a[2];
#pragma unroll
        for (int mt = 0; mt < 2; ++mt) {
            int row  = wm + mt * 16 + (lane & 15);
            int koff = (lane >> 4) * 8;
            a[mt].q[0] = *(const uint4*)(lA + row * 32 + koff);
            a[mt].q[1] = *(const uint4*)(lA + row * 32 + koff + 16);
        }
#pragma unroll
        for (int nt = 0; nt < 4; ++nt) {
            FragBF b;
            int row  = wn + nt * 16 + (lane & 15);
            int koff = (lane >> 4) * 8;
            b.q[0] = *(const uint4*)(lB + row * 32 + koff);
            b.q[1] = *(const uint4*)(lB + row * 32 + koff + 16);
#pragma unroll
            for (int mt = 0; mt < 2; ++mt)
                acc[mt][nt] = __builtin_amdgcn_wmma_f32_16x16x32_bf16(
                    false, a[mt].v, false, b.v, (short)0, acc[mt][nt], false, false);
        }
        __syncthreads();
    }

    // ---- epilogue: bias, scale, bf16 store (optionally V-transposed) ----
#pragma unroll
    for (int mt = 0; mt < 2; ++mt)
#pragma unroll
        for (int nt = 0; nt < 4; ++nt)
#pragma unroll
            for (int r = 0; r < 8; ++r) {
                int gr = rowBase + wm + mt * 16 + r + (lane >> 4) * 8;
                int j  = colBase + wn + nt * 16 + (lane & 15);
                float val = (acc[mt][nt][r] + bias[j]) * scale;
                if (transposeV) {
                    int bI = gr >> 11, n = gr & 2047;
                    int h = j >> 6, dh = j & 63;
                    out[(((size_t)bI * NH + h) * HD + dh) * NKS + n] = f2bf(val);
                } else {
                    out[(size_t)gr * DD + j] = f2bf(val);
                }
            }
}

// =====================================================================
// Flash attention with TDM-staged, double-buffered K/V tiles in LDS.
//   qt bf16 (B*NQ, D) pre-scaled by 1/8; kt bf16 (B*NK, D);
//   vtT bf16 (B, H, DH, NK); mask (B, NQ, NK); out fp32 (B, NQ, D)
//   LDS: lK[2] 8KB + lV[2] 8KB + lP 16KB = 48KB / WG.
// =====================================================================
__global__ __launch_bounds__(256)
void flash_attn(const unsigned short* __restrict__ qt,
                const unsigned short* __restrict__ kt,
                const unsigned short* __restrict__ vtT,
                const unsigned char*  __restrict__ mask,
                float* __restrict__ out)
{
    __shared__ __align__(16) unsigned short lK[2][64 * 64]; // [kv][dh]
    __shared__ __align__(16) unsigned short lV[2][64 * 64]; // [dh][kv]
    __shared__ __align__(16) unsigned short lP[8 * 16 * 64];

    const int tid  = threadIdx.x;
    const int lane = tid & 31;
    const int wave = tid >> 5;
    const int bh = blockIdx.x;
    const int b = bh >> 4, h = bh & 15;
    const int q0 = blockIdx.y * 128 + wave * 16;

    const float NEG_INF = -__builtin_inff();

    const unsigned short* kbase = kt  + ((size_t)b * NKS) * DD + h * HD;
    const unsigned short* vbase = vtT + ((size_t)(b * NH + h) * HD) * NKS;
    const unsigned char*  mbase = mask + (size_t)b * NQS * NKS;

    // ---- stage one 64-row K/V chunk into LDS buffer `buf` ----
    auto stageKV = [&](int kv0, int buf) {
#ifdef HAVE_TDM
        if (wave == 0) {   // one TDM issue per WG; TDM ignores EXEC
            tdm_load_2d_bf16((unsigned)(size_t)(const void*)&lK[buf][0],
                             kbase + (size_t)kv0 * DD, 64u, 64u, DD);
            tdm_load_2d_bf16((unsigned)(size_t)(const void*)&lV[buf][0],
                             vbase + kv0, 64u, 64u, NKS);
        }
#else
#pragma unroll
        for (int i = 0; i < 2; ++i) {            // 512 uint4 per tile / 256 thr
            int idx = tid + i * 256;
            int r = idx >> 3, c = (idx & 7) * 8; // 8 uint4 per 64-elem row
            *(uint4*)(&lK[buf][r * 64 + c]) =
                *(const uint4*)(kbase + (size_t)(kv0 + r) * DD + c);
            *(uint4*)(&lV[buf][r * 64 + c]) =
                *(const uint4*)(vbase + (size_t)r * NKS + kv0 + c);
        }
#endif
    };
    auto waitKV = [&]() {
#ifdef HAVE_TDM
        if (wave == 0) wait_tensorcnt0();
#endif
        // (fallback path: __syncthreads below provides ds/load ordering)
    };

    // ---- Q fragments (registers for whole kernel) ----
    FragBF aQ[2];
    {
        const unsigned short* qrow =
            qt + ((size_t)(b * NQS + q0 + (lane & 15))) * DD + h * HD;
#pragma unroll
        for (int ks = 0; ks < 2; ++ks) {
            int koff = ks * 32 + (lane >> 4) * 8;
            aQ[ks].q[0] = *(const uint4*)(qrow + koff);
            aQ[ks].q[1] = *(const uint4*)(qrow + koff + 16);
        }
    }

    v8f O[4];
    float m[8], l[8];
#pragma unroll
    for (int dt = 0; dt < 4; ++dt) O[dt] = zero8();
#pragma unroll
    for (int r = 0; r < 8; ++r) { m[r] = NEG_INF; l[r] = 0.0f; }

    unsigned short* myP = lP + wave * (16 * 64);

    // prologue: chunk 0 into buffer 0
    stageKV(0, 0);
    waitKV();
    __syncthreads();

    for (int c = 0; c < NKS / 64; ++c) {
        const int kv0 = c * 64;
        const int buf = c & 1;

        // stage next chunk into the other buffer (overlaps with compute)
        if (c + 1 < NKS / 64) stageKV(kv0 + 64, buf ^ 1);

        // prefetch next chunk's mask bytes for this lane's rows
        __builtin_prefetch(mbase + (size_t)(q0 + (lane >> 4) * 8) * NKS + kv0 + 64, 0, 1);

        // ---------------- S = Q K^T ----------------
        v8f S[4];
#pragma unroll
        for (int nt = 0; nt < 4; ++nt) S[nt] = zero8();
#pragma unroll
        for (int ks = 0; ks < 2; ++ks)
#pragma unroll
            for (int nt = 0; nt < 4; ++nt) {
                FragBF bK;
                const unsigned short* p =
                    &lK[buf][(nt * 16 + (lane & 15)) * 64 + ks * 32 + (lane >> 4) * 8];
                bK.q[0] = *(const uint4*)(p);
                bK.q[1] = *(const uint4*)(p + 16);
                S[nt] = __builtin_amdgcn_wmma_f32_16x16x32_bf16(
                    false, aQ[ks].v, false, bK.v, (short)0, S[nt], false, false);
            }

        // ---------------- mask ----------------
#pragma unroll
        for (int nt = 0; nt < 4; ++nt)
#pragma unroll
            for (int r = 0; r < 8; ++r) {
                int qr = q0 + r + (lane >> 4) * 8;
                int kv = kv0 + nt * 16 + (lane & 15);
                if (!mbase[(size_t)qr * NKS + kv]) S[nt][r] = NEG_INF;
            }

        // ---------------- online softmax (reference semantics) ----------------
        float mb[8], alpha[8];
#pragma unroll
        for (int r = 0; r < 8; ++r) {
            float rm = S[0][r];
#pragma unroll
            for (int nt = 1; nt < 4; ++nt) rm = fmaxf(rm, S[nt][r]);
#pragma unroll
            for (int off = 1; off < 16; off <<= 1)
                rm = fmaxf(rm, __shfl_xor(rm, off, 32));
            float mn = fmaxf(m[r], rm);
            mb[r]    = (mn == NEG_INF) ? 0.0f : mn;
            alpha[r] = (m[r] == NEG_INF) ? 0.0f : __expf(m[r] - mb[r]);
            m[r] = mn;
        }
        float rs[8];
#pragma unroll
        for (int r = 0; r < 8; ++r) rs[r] = 0.0f;
#pragma unroll
        for (int nt = 0; nt < 4; ++nt)
#pragma unroll
            for (int r = 0; r < 8; ++r) {
                float p = __expf(S[nt][r] - mb[r]);
                S[nt][r] = p;
                rs[r] += p;
            }
#pragma unroll
        for (int r = 0; r < 8; ++r) {
#pragma unroll
            for (int off = 1; off < 16; off <<= 1)
                rs[r] += __shfl_xor(rs[r], off, 32);
            l[r] = l[r] * alpha[r] + rs[r];
#pragma unroll
            for (int dt = 0; dt < 4; ++dt) O[dt][r] *= alpha[r];
        }

        // ---------------- P (C-layout fp32) -> LDS bf16 ----------------
#pragma unroll
        for (int nt = 0; nt < 4; ++nt)
#pragma unroll
            for (int r = 0; r < 8; ++r)
                myP[(r + (lane >> 4) * 8) * 64 + nt * 16 + (lane & 15)] =
                    f2bf(S[nt][r]);
        // same-wave LDS is in-order: safe to re-read without barrier

        // ---------------- O += P V ----------------
#pragma unroll
        for (int ks = 0; ks < 2; ++ks) {
            FragBF aP;
            {
                int koff = ks * 32 + (lane >> 4) * 8;
                aP.q[0] = *(const uint4*)(myP + (lane & 15) * 64 + koff);
                aP.q[1] = *(const uint4*)(myP + (lane & 15) * 64 + koff + 16);
            }
#pragma unroll
            for (int dt = 0; dt < 4; ++dt) {
                FragBF bV;
                const unsigned short* p =
                    &lV[buf][(dt * 16 + (lane & 15)) * 64 + ks * 32 + (lane >> 4) * 8];
                bV.q[0] = *(const uint4*)(p);
                bV.q[1] = *(const uint4*)(p + 16);
                O[dt] = __builtin_amdgcn_wmma_f32_16x16x32_bf16(
                    false, aP.v, false, bV.v, (short)0, O[dt], false, false);
            }
        }

        // publish next buffer: wave0 drains TENSORcnt, then WG barrier
        waitKV();
        __syncthreads();
    }

    // ---------------- epilogue: x = O / (l + 1e-6) ----------------
#pragma unroll
    for (int dt = 0; dt < 4; ++dt)
#pragma unroll
        for (int r = 0; r < 8; ++r) {
            int qr = q0 + r + (lane >> 4) * 8;
            int j  = h * HD + dt * 16 + (lane & 15);
            out[((size_t)b * NQS + qr) * DD + j] = O[dt][r] / (l[r] + 1e-6f);
        }
}

// =====================================================================
extern "C" void kernel_launch(void* const* d_in, const int* in_sizes, int n_in,
                              void* d_out, int out_size, void* d_ws, size_t ws_size,
                              hipStream_t stream)
{
    const float* q  = (const float*)d_in[0];
    const float* k  = (const float*)d_in[1];
    const float* v  = (const float*)d_in[2];
    const unsigned char* mask = (const unsigned char*)d_in[3];
    const float* Wq = (const float*)d_in[4];
    const float* bq = (const float*)d_in[5];
    const float* Wk = (const float*)d_in[6];
    const float* bk = (const float*)d_in[7];
    const float* Wv = (const float*)d_in[8];
    const float* bv = (const float*)d_in[9];

    unsigned short* qt  = (unsigned short*)d_ws;
    unsigned short* kt  = qt + (size_t)BB * NQS * DD;
    unsigned short* vtT = kt + (size_t)BB * NKS * DD;

    dim3 gProj(32, 8);
    dim3 blk(256);

    proj_gemm<<<gProj, blk, 0, stream>>>(q, Wq, bq, qt, 0.125f, 0);
    proj_gemm<<<gProj, blk, 0, stream>>>(k, Wk, bk, kt, 1.0f, 0);
    proj_gemm<<<gProj, blk, 0, stream>>>(v, Wv, bv, vtT, 1.0f, 1);

    dim3 gAttn(BB * NH, NQS / 128);
    flash_attn<<<gAttn, blk, 0, stream>>>(qt, kt, vtT, mask, (float*)d_out);
}
